// ColumnParallelFusedMoeLinear_41300405518714
// MI455X (gfx1250) — compile-verified, matrix operand order checked by reading, not run on hardware
//
#include <hip/hip_runtime.h>
#include <hip/hip_bf16.h>

// Problem constants (from reference)
#define TOK   8192
#define NEXP  8
#define DIN   1024
#define DOUT  2048

// Tiling
#define BM 128
#define BN 128
#define BK 32
#define LDF 36                 // padded LDS row stride in f32 elements (36 dwords -> 16 rows hit 16 distinct banks)
#define NTHREADS 256
#define ROW_TILES (TOK / BM)   // 64
#define TILEF (BM * LDF)       // floats per tile buffer (18432 B)

typedef __attribute__((ext_vector_type(16))) __bf16 v16bf;
typedef __attribute__((ext_vector_type(8)))  float  v8f;
typedef __attribute__((ext_vector_type(4)))  float  f4;
typedef unsigned int u32x4 __attribute__((ext_vector_type(4)));
typedef int          i32x4 __attribute__((ext_vector_type(4)));
typedef int          i32x8 __attribute__((ext_vector_type(8)));

#define AS1 __attribute__((address_space(1)))
#define AS3 __attribute__((address_space(3)))

// ---- data-mover path selection ------------------------------------------
#if __has_builtin(__builtin_amdgcn_tensor_load_to_lds)
#  define MOE_USE_TDM 1
#  if __has_include(<hip/amd_detail/amd_gfx1250_TDM.h>)
#    define MOE_TDM6 1          // amdgpu-toolchain: 6-arg builtin
#  endif
#elif __has_builtin(__builtin_amdgcn_global_load_async_to_lds_b128)
#  define MOE_USE_ASYNC 1
#endif

// Build a v16bf WMMA operand fragment from 8+8 contiguous f32 in LDS.
// CDNA5 16-bit operand layout: lane holds row (lane%16), K chunks at
// kbase=(lane/16)*8 (elems 0..7) and kbase+16 (elems 8..15).
__device__ inline v16bf pack_frag(const float* __restrict__ p) {
    f4 c0 = *reinterpret_cast<const f4*>(p);        // K +0..3
    f4 c1 = *reinterpret_cast<const f4*>(p + 4);    // K +4..7
    f4 c2 = *reinterpret_cast<const f4*>(p + 16);   // K +16..19
    f4 c3 = *reinterpret_cast<const f4*>(p + 20);   // K +20..23
    v16bf r;
    r[0]  = (__bf16)c0.x; r[1]  = (__bf16)c0.y; r[2]  = (__bf16)c0.z; r[3]  = (__bf16)c0.w;
    r[4]  = (__bf16)c1.x; r[5]  = (__bf16)c1.y; r[6]  = (__bf16)c1.z; r[7]  = (__bf16)c1.w;
    r[8]  = (__bf16)c2.x; r[9]  = (__bf16)c2.y; r[10] = (__bf16)c2.z; r[11] = (__bf16)c2.w;
    r[12] = (__bf16)c3.x; r[13] = (__bf16)c3.y; r[14] = (__bf16)c3.z; r[15] = (__bf16)c3.w;
    return r;
}

#if defined(MOE_USE_TDM)
// Issue one TDM 2D tile load: rows_valid rows of BK f32 each, row stride
// row_stride_elems (f32 units), into LDS at lds_off_bytes with pad_enable
// producing the 36-dword padded row stride (32 dwords data + 4 dwords pad).
__device__ inline void tdm_load_tile(unsigned lds_off_bytes,
                                     const float* __restrict__ gptr,
                                     int rows_valid,
                                     int row_stride_elems) {
    unsigned long long ga = (unsigned long long)(uintptr_t)gptr;
    u32x4 g0;
    g0.x = 1u;                                       // count=1: valid descriptor
    g0.y = lds_off_bytes;                            // lds_addr
    g0.z = (unsigned)(ga & 0xffffffffu);             // global_addr[31:0]
    g0.w = (unsigned)((ga >> 32) & 0x1ffffffu)       // global_addr[56:32]
         | (2u << 30);                               // type = 2 ("image")
    unsigned td0 = (unsigned)DIN;                    // tensor dim0 (elements)
    unsigned td1 = (unsigned)rows_valid;             // tensor dim1: OOB rows read 0
    i32x8 g1;
    g1[0] = (int)( (2u << 16)                        // data_size = 4B
                 | (1u << 20)                        // pad_enable
                 | (4u << 22)                        // pad_interval: 32 dwords
                 | (3u << 25) );                     // pad_amount: 4 dwords
    g1[1] = (int)((td0 & 0xffffu) << 16);            // tensor_dim0[15:0]
    g1[2] = (int)(((td0 >> 16) & 0xffffu)            // tensor_dim0[31:16]
                 | ((td1 & 0xffffu) << 16));         // tensor_dim1[15:0]
    g1[3] = (int)(((td1 >> 16) & 0xffffu)            // tensor_dim1[31:16]
                 | ((unsigned)BK << 16));            // tile_dim0 = 32
    g1[4] = (int)(BM & 0xffff);                      // tile_dim1 = 128, tile_dim2 = 0
    g1[5] = row_stride_elems;                        // tensor_dim0_stride[31:0]
    g1[6] = 0;
    g1[7] = 0;
    i32x4 z4 = {0, 0, 0, 0};
#if defined(MOE_TDM6)
    i32x8 z8 = {0, 0, 0, 0, 0, 0, 0, 0};
    __builtin_amdgcn_tensor_load_to_lds(g0, g1, z4, z4, z8, 0);
#else
    __builtin_amdgcn_tensor_load_to_lds(g0, g1, z4, z4, 0);
#endif
}
#endif

__global__ __launch_bounds__(NTHREADS)
void moe_grouped_gemm_bf16wmma(const float* __restrict__ x,
                               const float* __restrict__ w,
                               const int*   __restrict__ m_sizes,
                               float* __restrict__ y)
{
    const int tid  = threadIdx.x;
    const int lane = tid & 31;
    const int wid  = tid >> 5;       // 0..7
    const int wm   = wid & 1;        // 2 waves along M
    const int wn   = wid >> 1;       // 4 waves along N

    const int nblk = blockIdx.x;               // 0..DOUT/BN-1
    const int e    = blockIdx.y / ROW_TILES;
    const int rt   = blockIdx.y % ROW_TILES;

    int off = 0, me = 0;
#pragma unroll
    for (int i = 0; i < NEXP; ++i) {
        int s = m_sizes[i];
        off += (i < e) ? s : 0;
        me   = (i == e) ? s : me;
    }
    const int r0 = rt * BM;
    if (r0 >= me) return;            // uniform early exit

    const int n0 = nblk * BN;
    const float* wexp = w + (size_t)e * DOUT * DIN;

    // Dynamic LDS: A0 @0, A1 @TILEF, B0 @2*TILEF, B1 @3*TILEF (72 KB total).
    // NOTE: buffer pointers are always computed arithmetically from `buf`
    // (never stored in a braced pointer-array init) to avoid constant
    // addrspacecast aggregates that ld.lld rejects.
    extern __shared__ float smem[];

    // Zero-fill ragged-edge A rows once (both buffers); the data mover only
    // ever writes rows < me - r0, so these stay zero for every k-tile.
    for (int idx = tid; idx < TILEF; idx += NTHREADS) {
        int r = idx / LDF;
        if (r0 + r >= me) { smem[idx] = 0.f; smem[TILEF + idx] = 0.f; }
    }

    // ---- tile movers ----------------------------------------------------
#if defined(MOE_USE_TDM)
    auto move_tiles = [&](int buf, int k0) {
        if (wid == 0) {
            float* Ad = smem + buf * TILEF;
            float* Bd = smem + (2 + buf) * TILEF;
            tdm_load_tile((unsigned)(uintptr_t)(void*)Ad,
                          x + (size_t)(off + r0) * DIN + k0, me - r0, DIN);
            tdm_load_tile((unsigned)(uintptr_t)(void*)Bd,
                          wexp + (size_t)n0 * DIN + k0, DOUT - n0, DIN);
        }
    };
    auto wait_moves = [&]() {
        if (wid == 0) __builtin_amdgcn_s_wait_tensorcnt(0);
    };
#elif defined(MOE_USE_ASYNC)
    auto move_tiles = [&](int buf, int k0) {
        float* Ad = smem + buf * TILEF;
        float* Bd = smem + (2 + buf) * TILEF;
#pragma unroll
        for (int i = 0; i < 4; ++i) {
            const int idx = tid + i * NTHREADS;
            const int r   = idx >> 3;
            const int c   = (idx & 7) << 2;
            if ((r0 + r) < me)
                __builtin_amdgcn_global_load_async_to_lds_b128(
                    (AS1 void*)(void*)(x + (size_t)(off + r0 + r) * DIN + k0 + c),
                    (AS3 void*)(void*)(Ad + r * LDF + c), 0, 0);
            __builtin_amdgcn_global_load_async_to_lds_b128(
                (AS1 void*)(void*)(wexp + (size_t)(n0 + r) * DIN + k0 + c),
                (AS3 void*)(void*)(Bd + r * LDF + c), 0, 0);
        }
    };
    auto wait_moves = [&]() {
#if __has_builtin(__builtin_amdgcn_s_wait_asynccnt)
        __builtin_amdgcn_s_wait_asynccnt(0);
#else
        asm volatile("s_wait_asynccnt 0x0" ::: "memory");
#endif
    };
#else
    auto move_tiles = [&](int buf, int k0) {
        float* Ad = smem + buf * TILEF;
        float* Bd = smem + (2 + buf) * TILEF;
        const f4 fzero = {0.f, 0.f, 0.f, 0.f};
#pragma unroll
        for (int i = 0; i < 4; ++i) {
            const int idx = tid + i * NTHREADS;
            const int r   = idx >> 3;
            const int c   = (idx & 7) << 2;
            f4 va = fzero;
            if ((r0 + r) < me)
                va = *reinterpret_cast<const f4*>(x + (size_t)(off + r0 + r) * DIN + k0 + c);
            f4 vb = *reinterpret_cast<const f4*>(wexp + (size_t)(n0 + r) * DIN + k0 + c);
            *reinterpret_cast<f4*>(Ad + r * LDF + c) = va;
            *reinterpret_cast<f4*>(Bd + r * LDF + c) = vb;
        }
    };
    auto wait_moves = [&]() {};
#endif

    // ---- accumulators ---------------------------------------------------
    v8f acc[4][2];
    const v8f vz = {};
#pragma unroll
    for (int mt = 0; mt < 4; ++mt)
#pragma unroll
        for (int nt = 0; nt < 2; ++nt)
            acc[mt][nt] = vz;

    const int lrow = lane & 15;
    const int kgrp = (lane >> 4) << 3;   // 0 or 8: CDNA5 16-bit operand lane->K base

    auto compute = [&](int buf) {
        const float* Ab = smem + buf * TILEF;
        const float* Bb = smem + (2 + buf) * TILEF;
        v16bf a[4], b[2];
#pragma unroll
        for (int mt = 0; mt < 4; ++mt)
            a[mt] = pack_frag(Ab + (wm * 64 + mt * 16 + lrow) * LDF + kgrp);
#pragma unroll
        for (int nt = 0; nt < 2; ++nt)
            b[nt] = pack_frag(Bb + (wn * 32 + nt * 16 + lrow) * LDF + kgrp);
#pragma unroll
        for (int mt = 0; mt < 4; ++mt)
#pragma unroll
            for (int nt = 0; nt < 2; ++nt)
                acc[mt][nt] = __builtin_amdgcn_wmma_f32_16x16x32_bf16(
                    false, a[mt], false, b[nt],
                    (short)0, acc[mt][nt], false, false);
    };

    // ---- double-buffered pipeline ---------------------------------------
    const int NK = DIN / BK;   // 32 k-tiles
    move_tiles(0, 0);
    wait_moves();
    __syncthreads();

    for (int kt = 0; kt < NK; ++kt) {
        const int buf = kt & 1;
        if (kt + 1 < NK) move_tiles(buf ^ 1, (kt + 1) * BK);
        compute(buf);
        if (kt + 1 < NK) wait_moves();
        __syncthreads();
    }

    // ---- writeback: VGPR j of lane L holds (M = j + 8*(L/16), N = L%16) --
#pragma unroll
    for (int mt = 0; mt < 4; ++mt) {
#pragma unroll
        for (int nt = 0; nt < 2; ++nt) {
            const int n = n0 + wn * 32 + nt * 16 + lrow;
#pragma unroll
            for (int j = 0; j < 8; ++j) {
                const int rl = r0 + wm * 64 + mt * 16 + ((lane >> 4) << 3) + j;
                if (rl < me)
                    y[(size_t)(off + rl) * DOUT + n] = acc[mt][nt][j];
            }
        }
    }
}

extern "C" void kernel_launch(void* const* d_in, const int* in_sizes, int n_in,
                              void* d_out, int out_size, void* d_ws, size_t ws_size,
                              hipStream_t stream) {
    (void)in_sizes; (void)n_in; (void)out_size; (void)d_ws; (void)ws_size;
    const float* x       = (const float*)d_in[0];
    const float* w       = (const float*)d_in[1];
    const int*   m_sizes = (const int*)d_in[2];
    float*       y       = (float*)d_out;

    dim3 grid(DOUT / BN, NEXP * ROW_TILES);   // 16 x 512
    dim3 block(NTHREADS);
    const size_t shmem = 4u * TILEF * sizeof(float);   // 73728 B
    hipLaunchKernelGGL(moe_grouped_gemm_bf16wmma, grid, block, shmem, stream,
                       x, w, m_sizes, y);
}
